// GptLanguageModel_558345748671
// MI455X (gfx1250) — matrix operand order repaired and checked by reading, not compiled, not run on hardware
//
#include <hip/hip_runtime.h>
#include <hip/hip_bf16.h>

// MI455X / gfx1250, wave32. All GEMMs + attention on v_wmma_f32_16x16x32_bf16.
// Residual stream kept in fp32; bf16 mirrors feed the matrix units.
// GEMM: 256-thread WGs, 128x128 block tile, 32x64 wave tile, BK=64
//       (16 WMMA per wave per barrier pair), global_prefetch of next K tile,
//       ASYNC-to-LDS copies (global_load_async_to_lds_b128 + s_wait_asynccnt).
// Workspace requirement: ~175 MB (see layout in kernel_launch).

typedef __bf16 bf16_t;
typedef __attribute__((ext_vector_type(16))) __bf16 v16bf;
typedef __attribute__((ext_vector_type(8)))  __bf16 v8bf;
typedef __attribute__((ext_vector_type(8)))  float  v8f;
typedef __attribute__((ext_vector_type(4)))  int    v4i;

#define NB   4
#define NT   1024
#define NC   1024
#define NH   16
#define HD   64
#define NFF  4096
#define NV   32000
#define NROW (NB * NT)   // 4096

#if defined(__has_builtin)
#if __has_builtin(__builtin_amdgcn_global_load_async_to_lds_b128) && \
    __has_builtin(__builtin_amdgcn_s_wait_asynccnt)
#define USE_ASYNC_LDS 1
#endif
#endif

#define AS1 __attribute__((address_space(1)))
#define AS3 __attribute__((address_space(3)))

// ---------- helpers ----------
__device__ __forceinline__ bf16_t f2bf(float f) {
  unsigned u = __builtin_bit_cast(unsigned, f);
  u += 0x7FFFu + ((u >> 16) & 1u);              // round-to-nearest-even
  return __builtin_bit_cast(bf16_t, (unsigned short)(u >> 16));
}

__device__ __forceinline__ v8f wmma_bf16(v16bf a, v16bf b, v8f c) {
  return __builtin_amdgcn_wmma_f32_16x16x32_bf16(
      false, a, false, b, (short)0, c, false, false);
}

// 16B copy global->LDS: async if available, else load+store.
__device__ __forceinline__ void cp16(bf16_t* lds, const bf16_t* g) {
#ifdef USE_ASYNC_LDS
  __builtin_amdgcn_global_load_async_to_lds_b128(
      (AS1 v4i*)(AS1 void*)(void*)g, (AS3 v4i*)(AS3 void*)lds, 0, 0);
#else
  *(v8bf*)lds = *(const v8bf*)g;
#endif
}
__device__ __forceinline__ void cp_fence() {
#ifdef USE_ASYNC_LDS
  __builtin_amdgcn_s_wait_asynccnt(0);
#endif
}

// A fragment (16x32 bf16, M x K). tile is row-major, stride lda elements.
// ISA 7.12.2: lanes 0-15 row M=lane, K={0..7,16..23}; lanes 16-31 row M=lane-16,
// K={8..15,24..31}.
__device__ __forceinline__ v16bf load_frag_a(const bf16_t* tile, int mbase,
                                             int lda, int kofs) {
  int lane = threadIdx.x & 31;
  int r = lane & 15, half = lane >> 4;
  const bf16_t* p = tile + (size_t)(mbase + r) * lda + kofs + half * 8;
  v8bf lo = *(const v8bf*)(p);
  v8bf hi = *(const v8bf*)(p + 16);
  v16bf f;
#pragma unroll
  for (int i = 0; i < 8; ++i) { f[i] = lo[i]; f[i + 8] = hi[i]; }
  return f;
}

// B fragment (32x16 bf16, K x N) read from an N x K (row-major) LDS tile.
// ISA: lane = column N (mod 16); lanes 0-15 hold K={0..7,16..23}, lanes 16-31 K={8..15,24..31}.
__device__ __forceinline__ v16bf load_frag_b(const bf16_t* tileNxK, int nbase,
                                             int ldb, int kofs) {
  int lane = threadIdx.x & 31;
  int n = lane & 15, half = lane >> 4;
  const bf16_t* p = tileNxK + (size_t)(nbase + n) * ldb + kofs + half * 8;
  v8bf lo = *(const v8bf*)(p);
  v8bf hi = *(const v8bf*)(p + 16);
  v16bf f;
#pragma unroll
  for (int i = 0; i < 8; ++i) { f[i] = lo[i]; f[i + 8] = hi[i]; }
  return f;
}

// ---------- prep kernels ----------
__global__ __launch_bounds__(256) void embed_kernel(
    const int* __restrict__ idx, const float* __restrict__ temb,
    const float* __restrict__ pemb, float* __restrict__ x,
    bf16_t* __restrict__ xb) {
  int row = blockIdx.x;            // 0..4095
  int t = row & (NT - 1);
  int tok = idx[row];
  for (int c = threadIdx.x; c < NC; c += 256) {
    float v = temb[(size_t)tok * NC + c] + pemb[(size_t)t * NC + c];
    x[(size_t)row * NC + c] = v;
    xb[(size_t)row * NC + c] = f2bf(v);
  }
}

__global__ __launch_bounds__(256) void convert_to_bf16(
    const float* __restrict__ in, bf16_t* __restrict__ out, long n) {
  long i = (long)blockIdx.x * 256 + threadIdx.x;
  long stride = (long)gridDim.x * 256;
  for (; i < n; i += stride) out[i] = f2bf(in[i]);
}

// in: K x N fp32 (row-major) -> out: N x K bf16 (row-major)
__global__ __launch_bounds__(256) void transpose_to_bf16(
    const float* __restrict__ in, bf16_t* __restrict__ out, int K, int N) {
  __shared__ float tile[32][33];
  int kb = blockIdx.x * 32, nb = blockIdx.y * 32;
  int tx = threadIdx.x & 31, ty = threadIdx.x >> 5;  // 32 x 8
#pragma unroll
  for (int i = 0; i < 32; i += 8)
    tile[ty + i][tx] = in[(size_t)(kb + ty + i) * N + nb + tx];
  __syncthreads();
#pragma unroll
  for (int i = 0; i < 32; i += 8)
    out[(size_t)(nb + ty + i) * K + kb + tx] = f2bf(tile[tx][ty + i]);
}

// ---------- WMMA GEMM: C(MxN) = A(MxK,bf16) * Bt(NxK,bf16)^T ----------
// Block tile 128x128, 8 waves, wave tile 32x64, BK=64.
enum { EPI_BF16 = 0, EPI_RELU = 1, EPI_RESID = 2, EPI_F32 = 3 };

template <int EPI>
__global__ __launch_bounds__(256) void gemm_wmma(
    const bf16_t* __restrict__ A, const bf16_t* __restrict__ Bt,
    const float* resid, float* outF, bf16_t* outB, int M, int N, int K) {
  (void)M;
  __shared__ __align__(16) bf16_t sA[128 * 64];  // M x K tile (16 KB)
  __shared__ __align__(16) bf16_t sB[128 * 64];  // N x K tile (16 KB)
  int wgM = blockIdx.x * 128, wgN = blockIdx.y * 128;
  int tid = threadIdx.x, wave = tid >> 5, lane = tid & 31;
  int waveM = (wave & 3) * 32, waveN = (wave >> 2) * 64;
  v8f acc[2][4] = {};
  int lrow = tid >> 1, lcol = (tid & 1) * 32;    // 128 rows x 64 cols, 2 thr/row
  for (int k0 = 0; k0 < K; k0 += 64) {
    const bf16_t* ga = A + (size_t)(wgM + lrow) * K + k0 + lcol;
    const bf16_t* gb = Bt + (size_t)(wgN + lrow) * K + k0 + lcol;
#pragma unroll
    for (int i = 0; i < 32; i += 8) {
      cp16(sA + lrow * 64 + lcol + i, ga + i);
      cp16(sB + lrow * 64 + lcol + i, gb + i);
    }
    if (k0 + 64 < K) {                     // pull next K tiles toward the WGP
      __builtin_prefetch(ga + 64, 0, 1);
      __builtin_prefetch(gb + 64, 0, 1);
    }
    cp_fence();
    __syncthreads();
#pragma unroll
    for (int kk = 0; kk < 64; kk += 32) {
      v16bf a0 = load_frag_a(sA, waveM + 0, 64, kk);
      v16bf a1 = load_frag_a(sA, waveM + 16, 64, kk);
#pragma unroll
      for (int ni = 0; ni < 4; ++ni) {
        v16bf b = load_frag_b(sB, waveN + ni * 16, 64, kk);
        acc[0][ni] = wmma_bf16(a0, b, acc[0][ni]);
        acc[1][ni] = wmma_bf16(a1, b, acc[1][ni]);
      }
    }
    __syncthreads();
  }
  // C layout: VGPR e, lanes 0-15 -> M=e, N=lane; lanes 16-31 -> M=e+8, N=lane-16
  int nl = lane & 15, hoff = (lane >> 4) * 8;
#pragma unroll
  for (int mi = 0; mi < 2; ++mi)
#pragma unroll
    for (int ni = 0; ni < 4; ++ni)
#pragma unroll
      for (int e = 0; e < 8; ++e) {
        size_t row = wgM + waveM + mi * 16 + e + hoff;
        size_t col = wgN + waveN + ni * 16 + nl;
        float v = acc[mi][ni][e];
        if (EPI == EPI_RELU) v = fmaxf(v, 0.0f);
        if (EPI == EPI_RESID) v += resid[row * N + col];
        if (EPI == EPI_RESID || EPI == EPI_F32) outF[row * N + col] = v;
        if (EPI != EPI_F32) outB[row * N + col] = f2bf(v);
      }
}

// ---------- flash attention (causal), WMMA for S=Q*K^T and O+=P*V ----------
// Q,K,V,O: (B*T) x 1024 bf16, column = head*64 + hd. grid=(T/64, B*NH), 128 thr.
__global__ __launch_bounds__(128) void flash_attn(
    const bf16_t* __restrict__ Q, const bf16_t* __restrict__ Km,
    const bf16_t* __restrict__ V, bf16_t* __restrict__ O) {
  __shared__ __align__(16) bf16_t sQ[64 * 64];     // q-rows x hd
  __shared__ __align__(16) bf16_t sK[32 * 64];     // keys  x hd   (N x K for S)
  __shared__ __align__(16) bf16_t sVt[64 * 32];    // hd    x keys (N x K for O)
  __shared__ __align__(16) bf16_t sP[4][16 * 32];  // per-wave P staging (M x K)

  int qb = blockIdx.x, bn = blockIdx.y;
  int b = bn >> 4, h = bn & 15;
  int colbase = h * HD;
  size_t rowbase = (size_t)b * NT;
  int tid = threadIdx.x, wave = tid >> 5, lane = tid & 31;
  int nl = lane & 15, hoff = (lane >> 4) * 8;
  int qrow0 = qb * 64 + wave * 16;  // this wave's first q index (within seq)

  // load Q tile 64x64
  {
    int r = tid >> 1, cs = (tid & 1) * 32;
    const bf16_t* g = Q + (rowbase + qb * 64 + r) * NC + colbase + cs;
#pragma unroll
    for (int i = 0; i < 32; i += 8)
      *(v8bf*)(sQ + r * 64 + cs + i) = *(const v8bf*)(g + i);
  }

  float mrow[8], lrow[8];
  v8f o[4] = {};
#pragma unroll
  for (int e = 0; e < 8; ++e) { mrow[e] = -1e30f; lrow[e] = 0.0f; }

  int kend = (qb + 1) * 64;
  for (int kb = 0; kb < kend; kb += 32) {
    __syncthreads();  // protect sK/sVt reuse (also orders first Q load)
    {   // K tile: 32 keys x 64 hd, direct copy
      int r = tid >> 2, cs = (tid & 3) * 16;
      const bf16_t* g = Km + (rowbase + kb + r) * NC + colbase + cs;
      *(v8bf*)(sK + r * 64 + cs)     = *(const v8bf*)(g);
      *(v8bf*)(sK + r * 64 + cs + 8) = *(const v8bf*)(g + 8);
      if (kb + 32 < kend) __builtin_prefetch(g + 32 * NC, 0, 1);
    }
    {   // V tile transposed into hd x keys
      int key = tid >> 2, cs = (tid & 3) * 16;
      const bf16_t* g = V + (rowbase + kb + key) * NC + colbase + cs;
#pragma unroll
      for (int j = 0; j < 16; ++j) sVt[(cs + j) * 32 + key] = g[j];
      if (kb + 32 < kend) __builtin_prefetch(g + 32 * NC, 0, 1);
    }
    __syncthreads();

    // S = Q * K^T : two 16x16 fragments (keys 0-15, 16-31), K-dim = 64
    v8f s0 = {}, s1 = {};
#pragma unroll
    for (int kk = 0; kk < 64; kk += 32) {
      v16bf aq = load_frag_a(sQ, wave * 16, 64, kk);
      s0 = wmma_bf16(aq, load_frag_b(sK, 0, 64, kk), s0);
      s1 = wmma_bf16(aq, load_frag_b(sK, 16, 64, kk), s1);
    }

    float sc[2][8];
#pragma unroll
    for (int e = 0; e < 8; ++e) {
      int qr = qrow0 + e + hoff;
      float v0 = s0[e] * 0.125f, v1 = s1[e] * 0.125f;   // HEAD_DIM^-0.5
      if (kb + nl > qr)      v0 = -1e30f;                // causal mask
      if (kb + 16 + nl > qr) v1 = -1e30f;
      sc[0][e] = v0; sc[1][e] = v1;
      // row max across the 16 lanes holding this row
      float rm = fmaxf(v0, v1);
#pragma unroll
      for (int msk = 8; msk >= 1; msk >>= 1)
        rm = fmaxf(rm, __shfl_xor(rm, msk, 32));
      float mn = fmaxf(mrow[e], rm);
      float alpha = __expf(mrow[e] - mn);
      float p0 = __expf(v0 - mn), p1 = __expf(v1 - mn);
      sc[0][e] = p0; sc[1][e] = p1;
      float rs = p0 + p1;
#pragma unroll
      for (int msk = 8; msk >= 1; msk >>= 1)
        rs += __shfl_xor(rs, msk, 32);
      lrow[e] = lrow[e] * alpha + rs;
      mrow[e] = mn;
#pragma unroll
      for (int nt = 0; nt < 4; ++nt) o[nt][e] *= alpha;
    }
    // stage P (16 q-rows x 32 keys) into wave-private LDS, reload as A frag
#pragma unroll
    for (int e = 0; e < 8; ++e) {
      sP[wave][(e + hoff) * 32 + nl]      = f2bf(sc[0][e]);
      sP[wave][(e + hoff) * 32 + 16 + nl] = f2bf(sc[1][e]);
    }
    v16bf ap = load_frag_a(sP[wave], 0, 32, 0);
#pragma unroll
    for (int nt = 0; nt < 4; ++nt)
      o[nt] = wmma_bf16(ap, load_frag_b(sVt, nt * 16, 32, 0), o[nt]);
  }

  // normalize + store O (bf16, same (B*T) x 1024 layout)
#pragma unroll
  for (int nt = 0; nt < 4; ++nt)
#pragma unroll
    for (int e = 0; e < 8; ++e) {
      size_t row = rowbase + qrow0 + e + hoff;
      size_t col = colbase + nt * 16 + nl;
      O[row * NC + col] = f2bf(o[nt][e] / lrow[e]);
    }
}

// ---------- host ----------
extern "C" void kernel_launch(void* const* d_in, const int* in_sizes, int n_in,
                              void* d_out, int out_size, void* d_ws,
                              size_t ws_size, hipStream_t stream) {
  (void)in_sizes; (void)n_in; (void)out_size; (void)ws_size;
  const int*   idx   = (const int*)d_in[0];
  const float* temb  = (const float*)d_in[1];
  const float* pemb  = (const float*)d_in[2];
  const float* wk    = (const float*)d_in[3];
  const float* wq    = (const float*)d_in[4];
  const float* wv    = (const float*)d_in[5];
  const float* wproj = (const float*)d_in[6];
  const float* win   = (const float*)d_in[7];
  const float* wout  = (const float*)d_in[8];
  float* logits = (float*)d_out;

  char* ws = (char*)d_ws;
  size_t off = 0;
  auto carve = [&](size_t bytes) {
    void* p = ws + off; off += (bytes + 255) & ~(size_t)255; return p;
  };
  float*  x     = (float*) carve((size_t)NROW * NC * 4);   // fp32 residual
  bf16_t* xb    = (bf16_t*)carve((size_t)NROW * NC * 2);   // bf16 mirror
  bf16_t* kbuf  = (bf16_t*)carve((size_t)NROW * NC * 2);
  bf16_t* qbuf  = (bf16_t*)carve((size_t)NROW * NC * 2);
  bf16_t* vbuf  = (bf16_t*)carve((size_t)NROW * NC * 2);
  bf16_t* obuf  = (bf16_t*)carve((size_t)NROW * NC * 2);
  bf16_t* hbuf  = (bf16_t*)carve((size_t)NROW * NFF * 2);
  bf16_t* wkT   = (bf16_t*)carve((size_t)NC * NC * 2);
  bf16_t* wqT   = (bf16_t*)carve((size_t)NC * NC * 2);
  bf16_t* wvT   = (bf16_t*)carve((size_t)NC * NC * 2);
  bf16_t* wpB   = (bf16_t*)carve((size_t)NC * NC * 2);     // w_proj as N x K
  bf16_t* winT  = (bf16_t*)carve((size_t)NC * NFF * 2);
  bf16_t* woutT = (bf16_t*)carve((size_t)NFF * NC * 2);
  bf16_t* tembB = (bf16_t*)carve((size_t)NV * NC * 2);     // head B, N x K

  // prep
  embed_kernel<<<NROW, 256, 0, stream>>>(idx, temb, pemb, x, xb);
  transpose_to_bf16<<<dim3(NC / 32, NC / 32), 256, 0, stream>>>(wk, wkT, NC, NC);
  transpose_to_bf16<<<dim3(NC / 32, NC / 32), 256, 0, stream>>>(wq, wqT, NC, NC);
  transpose_to_bf16<<<dim3(NC / 32, NC / 32), 256, 0, stream>>>(wv, wvT, NC, NC);
  convert_to_bf16<<<2048, 256, 0, stream>>>(wproj, wpB, (long)NC * NC);
  transpose_to_bf16<<<dim3(NC / 32, NFF / 32), 256, 0, stream>>>(win, winT, NC, NFF);
  transpose_to_bf16<<<dim3(NFF / 32, NC / 32), 256, 0, stream>>>(wout, woutT, NFF, NC);
  convert_to_bf16<<<8192, 256, 0, stream>>>(temb, tembB, (long)NV * NC);

  dim3 gN1(NROW / 128, NC / 128);     // (32, 8)
  dim3 gFF(NROW / 128, NFF / 128);    // (32, 32)
  for (int layer = 0; layer < 4; ++layer) {
    // source quirk: k <- wk, v <- wq, q <- wv
    gemm_wmma<EPI_BF16><<<gN1, 256, 0, stream>>>(xb, wkT, nullptr, nullptr, kbuf, NROW, NC, NC);
    gemm_wmma<EPI_BF16><<<gN1, 256, 0, stream>>>(xb, wqT, nullptr, nullptr, vbuf, NROW, NC, NC);
    gemm_wmma<EPI_BF16><<<gN1, 256, 0, stream>>>(xb, wvT, nullptr, nullptr, qbuf, NROW, NC, NC);
    flash_attn<<<dim3(NT / 64, NB * NH), 128, 0, stream>>>(qbuf, kbuf, vbuf, obuf);
    // out-proj + residual:  B[N=c,K=nh] = w_proj[c,nh] (already N x K)
    gemm_wmma<EPI_RESID><<<gN1, 256, 0, stream>>>(obuf, wpB, x, x, xb, NROW, NC, NC);
    // FFN
    gemm_wmma<EPI_RELU><<<gFF, 256, 0, stream>>>(xb, winT, nullptr, nullptr, hbuf, NROW, NFF, NC);
    gemm_wmma<EPI_RESID><<<gN1, 256, 0, stream>>>(hbuf, woutT, x, x, xb, NROW, NC, NFF);
  }
  // tied head: logits = x @ token_emb^T ; token_emb is already N x K
  gemm_wmma<EPI_F32><<<dim3(NROW / 128, NV / 128), 256, 0, stream>>>(
      xb, tembB, nullptr, logits, nullptr, NROW, NV, NC);
}
